// MultiHeadAttention_71064528880296
// MI455X (gfx1250) — compile-verified, matrix operand order checked by reading, not run on hardware
//
#include <hip/hip_runtime.h>
#include <hip/hip_bf16.h>

#define DM   4096
#define NH   32
#define DH   128
#define SEQL 2048
#define KSTEP 64

typedef __attribute__((ext_vector_type(16))) __bf16 v16bf;
typedef __attribute__((ext_vector_type(8)))  __bf16 v8bf;
typedef __attribute__((ext_vector_type(8)))  float  v8f;
typedef unsigned int v4u  __attribute__((ext_vector_type(4)));
typedef int          v8i_ __attribute__((ext_vector_type(8)));
typedef int          v4i_ __attribute__((ext_vector_type(4)));

#if defined(__gfx1250__) && __has_builtin(__builtin_amdgcn_tensor_load_to_lds)
#define USE_TDM 1
#else
#define USE_TDM 0
#endif

__device__ __forceinline__ __bf16 f2bf(float f) {
    unsigned u = __builtin_bit_cast(unsigned, f);
    unsigned r = u + 0x7FFFu + ((u >> 16) & 1u);   // round-to-nearest-even
    unsigned short h = (unsigned short)(r >> 16);
    return __builtin_bit_cast(__bf16, h);
}

__device__ __forceinline__ v8f wmma_bf16(v16bf a, v16bf b, v8f c) {
    return __builtin_amdgcn_wmma_f32_16x16x32_bf16(
        false, a, false, b, (short)0, c, false, false);
}

// 16x32 bf16 A-fragment (or 32x16 B-fragment by symmetry) from row-major
// [row][ld]: lane L -> row (L&15), K-chunks at ((L>>4)*8) and +16.
__device__ __forceinline__ v16bf load_frag(const __bf16* base, int row, int ld,
                                           int k0, int lane) {
    int r  = row + (lane & 15);
    int kk = k0 + ((lane >> 4) << 3);
    const __bf16* p = base + (size_t)r * ld + kk;
    v8bf c0 = *(const v8bf*)(p);
    v8bf c1 = *(const v8bf*)(p + 16);
    return __builtin_shufflevector(c0, c1, 0,1,2,3,4,5,6,7,8,9,10,11,12,13,14,15);
}

#if USE_TDM
// Issue a 2D TDM load: tile (tile_d1 rows x tile_d0 cols) of bf16 from
// row-major global (row stride = stride0 elements) into LDS at lds_off.
// D# packing per CDNA5 ISA (group0/group1).
// 6-arg builtin: (uint32x4 g0, int32x8 g1, int32x4, int32x4, int32x8, i32 cpol)
__device__ __forceinline__ void tdm_load_tile(const __bf16* gsrc, unsigned lds_off,
                                              unsigned tensor_d0, unsigned tensor_d1,
                                              unsigned tile_d0, unsigned tile_d1,
                                              unsigned long long stride0) {
    unsigned long long ga = (unsigned long long)(uintptr_t)gsrc;
    v4u g0;
    g0[0] = 1u;                                        // count=1, user D#
    g0[1] = lds_off;                                   // lds_addr (bytes)
    g0[2] = (unsigned)(ga & 0xFFFFFFFFu);              // global_addr[31:0]
    g0[3] = (unsigned)((ga >> 32) & 0x01FFFFFFu)       // global_addr[56:32]
          | 0x80000000u;                               // type=2 ("image")
    v8i_ g1;
    g1[0] = (int)(1u << 16);                           // wg_mask=0, data_size=1 (2B)
    g1[1] = (int)((tensor_d0 & 0xFFFFu) << 16);        // tensor_dim0[15:0]
    g1[2] = (int)((tensor_d0 >> 16) | ((tensor_d1 & 0xFFFFu) << 16));
    g1[3] = (int)((tensor_d1 >> 16) | (tile_d0 << 16)); // tile_dim0
    g1[4] = (int)(tile_d1 & 0xFFFFu);                  // tile_dim1, tile_dim2=0
    g1[5] = (int)(stride0 & 0xFFFFFFFFu);              // tensor_dim0_stride[31:0]
    g1[6] = (int)((stride0 >> 32) & 0xFFFFu);          // stride0[47:32], stride1=0
    g1[7] = 0;
    v4i_ gz4;
    gz4[0] = 0; gz4[1] = 0; gz4[2] = 0; gz4[3] = 0;
    v8i_ gz8;
    gz8[0] = 0; gz8[1] = 0; gz8[2] = 0; gz8[3] = 0;
    gz8[4] = 0; gz8[5] = 0; gz8[6] = 0; gz8[7] = 0;
    __builtin_amdgcn_tensor_load_to_lds(g0, g1, gz4, gz4, gz8, 0);
}
#endif

// ---------------- fp32 -> bf16 convert ----------------
__global__ void cvt_bf16_kernel(const float* __restrict__ in,
                                __bf16* __restrict__ out, int n) {
    int tid = blockIdx.x * blockDim.x + threadIdx.x;
    if (tid < n) out[tid] = f2bf(in[tid]);
}

// ---------------- bf16 GEMM:  C[M][N] = A[M][K] * W[N][K]^T + bias ----------------
__global__ __launch_bounds__(256) void gemm_bf16_kernel(
    const __bf16* __restrict__ A, const __bf16* __restrict__ W,
    const float* __restrict__ bias, float* __restrict__ C,
    int M, int N, int K) {
    __shared__ alignas(16) __bf16 tA[2][128 * KSTEP];
    __shared__ alignas(16) __bf16 tB[2][128 * KSTEP];

    int lane = threadIdx.x & 31;
    int w    = threadIdx.x >> 5;
    int wr   = w >> 1;          // 0..3 -> 32-row strip
    int wc   = w & 1;           // 0..1 -> 64-col strip
    int row0 = blockIdx.y * 128;
    int col0 = blockIdx.x * 128;

    v8f acc[2][4];
#pragma unroll
    for (int i = 0; i < 2; ++i)
#pragma unroll
        for (int j = 0; j < 4; ++j)
#pragma unroll
            for (int e = 0; e < 8; ++e) acc[i][j][e] = 0.0f;

#if USE_TDM
    if (w == 0) {
        tdm_load_tile(A + (size_t)row0 * K, (unsigned)(uintptr_t)&tA[0][0],
                      (unsigned)K, (unsigned)(M - row0), KSTEP, 128, (unsigned long long)K);
        tdm_load_tile(W + (size_t)col0 * K, (unsigned)(uintptr_t)&tB[0][0],
                      (unsigned)K, (unsigned)(N - col0), KSTEP, 128, (unsigned long long)K);
    }
#endif

    int buf = 0;
    for (int kt = 0; kt < K; kt += KSTEP) {
#if USE_TDM
        bool hasNext = (kt + KSTEP) < K;
        if (w == 0) {
            if (hasNext) {
                tdm_load_tile(A + (size_t)row0 * K + kt + KSTEP,
                              (unsigned)(uintptr_t)&tA[buf ^ 1][0],
                              (unsigned)(K - kt - KSTEP), (unsigned)(M - row0),
                              KSTEP, 128, (unsigned long long)K);
                tdm_load_tile(W + (size_t)col0 * K + kt + KSTEP,
                              (unsigned)(uintptr_t)&tB[buf ^ 1][0],
                              (unsigned)(K - kt - KSTEP), (unsigned)(N - col0),
                              KSTEP, 128, (unsigned long long)K);
                __builtin_amdgcn_s_wait_tensorcnt(2);   // current tiles done
            } else {
                __builtin_amdgcn_s_wait_tensorcnt(0);
            }
        }
        __syncthreads();
#else
        {
            int t = threadIdx.x;
#pragma unroll
            for (int c = 0; c < (128 * KSTEP) / (256 * 8); ++c) {
                int e  = (c * 256 + t) * 8;
                int r  = e / KSTEP;
                int cc = e % KSTEP;
                *(v8bf*)&tA[buf][e] = *(const v8bf*)(A + (size_t)(row0 + r) * K + kt + cc);
                *(v8bf*)&tB[buf][e] = *(const v8bf*)(W + (size_t)(col0 + r) * K + kt + cc);
            }
        }
        __syncthreads();
#endif
        const __bf16* At = &tA[buf][0];
        const __bf16* Bt = &tB[buf][0];
#pragma unroll
        for (int kc = 0; kc < KSTEP / 32; ++kc) {
            v16bf a0 = load_frag(At, wr * 32,      KSTEP, kc * 32, lane);
            v16bf a1 = load_frag(At, wr * 32 + 16, KSTEP, kc * 32, lane);
            v16bf b[4];
#pragma unroll
            for (int j = 0; j < 4; ++j)
                b[j] = load_frag(Bt, wc * 64 + 16 * j, KSTEP, kc * 32, lane);
#pragma unroll
            for (int j = 0; j < 4; ++j) {
                acc[0][j] = wmma_bf16(a0, b[j], acc[0][j]);
                acc[1][j] = wmma_bf16(a1, b[j], acc[1][j]);
            }
        }
        __syncthreads();
        buf ^= 1;
    }

    int hf = lane >> 4;
    int nc = lane & 15;
    int m0 = row0 + wr * 32;
    int n0 = col0 + wc * 64;
#pragma unroll
    for (int i = 0; i < 2; ++i)
#pragma unroll
        for (int j = 0; j < 4; ++j) {
            int n = n0 + j * 16 + nc;
            float bb = bias[n];
#pragma unroll
            for (int e = 0; e < 8; ++e) {
                int m = m0 + i * 16 + e + 8 * hf;
                C[(size_t)m * N + n] = acc[i][j][e] + bb;
            }
        }
}

// ---------------- RoPE + head-major bf16 pack:  P[s][h*DH+d] -> out[h][s][d] ----------------
__global__ void rope_pack_kernel(const float* __restrict__ P, __bf16* __restrict__ out) {
    int tid = blockIdx.x * blockDim.x + threadIdx.x;   // SEQL*DM/2 threads
    int s = tid / (DM / 2);
    int r = tid % (DM / 2);
    int h = r / (DH / 2);
    int i = r % (DH / 2);
    float theta = __expf(-((float)(2 * i) / (float)DH) * 9.210340371976184f); // ln(1e4)
    float ang = (float)s * theta;
    float cs = __cosf(ang), sn = __sinf(ang);
    size_t ib = (size_t)s * DM + h * DH + 2 * i;
    float e = P[ib], o = P[ib + 1];
    size_t ob = ((size_t)h * SEQL + s) * DH + 2 * i;
    out[ob]     = f2bf(e * cs - o * sn);
    out[ob + 1] = f2bf(e * sn + o * cs);
}

// ---------------- V pack transposed:  P[s][h*DH+d] -> out[h][d][s] ----------------
__global__ void vt_pack_kernel(const float* __restrict__ P, __bf16* __restrict__ out) {
    int tid = blockIdx.x * blockDim.x + threadIdx.x;   // SEQL*DM threads
    int s = tid / DM;
    int c = tid % DM;
    int h = c / DH, d = c % DH;
    out[((size_t)h * DH + d) * SEQL + s] = f2bf(P[(size_t)s * DM + c]);
}

// ---------------- causal flash attention, TDM-staged K/V ----------------
// grid (SEQL/128, NH), block 256.  Wave w owns the 16-query tile
// qs = bx*128 + w*16.  Key loop is block-uniform (ktEnd = (bx+1)*128) so K/V
// tiles can be double-buffered through LDS via TDM; each wave skips compute
// for key blocks entirely above its causal range (wave-uniform branch,
// barriers stay block-uniform).
__global__ __launch_bounds__(256) void flash_attn_kernel(
    const __bf16* __restrict__ Qb, const __bf16* __restrict__ Kb,
    const __bf16* __restrict__ Vt, __bf16* __restrict__ Ob) {
    __shared__ alignas(16) __bf16 tK[2][32 * DH];   // keys x d   (8 KB each)
    __shared__ alignas(16) __bf16 tV[2][DH * 32];   // d   x keys (8 KB each)
    __shared__ alignas(16) __bf16 lds_p[8][16 * 32];

    int lane = threadIdx.x & 31;
    int w    = threadIdx.x >> 5;
    int h    = blockIdx.y;
    int qs   = blockIdx.x * 128 + w * 16;
    int ktEnd = (blockIdx.x + 1) * 128;
    const float scale = 0.08838834764831845f;   // 1/sqrt(128)

    const __bf16* Qh = Qb + (size_t)h * SEQL * DH;
    const __bf16* Kh = Kb + (size_t)h * SEQL * DH;
    const __bf16* Vh = Vt + (size_t)h * DH * SEQL;

    v16bf aq[4];
#pragma unroll
    for (int c = 0; c < 4; ++c) aq[c] = load_frag(Qh, qs, DH, c * 32, lane);

    v8f o[8];
#pragma unroll
    for (int j = 0; j < 8; ++j)
#pragma unroll
        for (int e = 0; e < 8; ++e) o[j][e] = 0.0f;

    float mi[8], li[8];
#pragma unroll
    for (int i = 0; i < 8; ++i) { mi[i] = -3.0e38f; li[i] = 0.0f; }

    int hf = lane >> 4;
    int nc = lane & 15;

#if USE_TDM
    if (w == 0) {
        // K tile: 32 key-rows x 128 d from Kb[h][kt..][.]
        tdm_load_tile(Kh, (unsigned)(uintptr_t)&tK[0][0],
                      DH, SEQL, DH, 32, DH);
        // V tile: 128 d-rows x 32 keys from Vt[h][.][kt..]
        tdm_load_tile(Vh, (unsigned)(uintptr_t)&tV[0][0],
                      SEQL, DH, 32, DH, SEQL);
    }
#endif

    int buf = 0;
    for (int kt = 0; kt < ktEnd; kt += 32) {
#if USE_TDM
        bool hasNext = (kt + 32) < ktEnd;
        if (w == 0) {
            if (hasNext) {
                tdm_load_tile(Kh + (size_t)(kt + 32) * DH,
                              (unsigned)(uintptr_t)&tK[buf ^ 1][0],
                              DH, (unsigned)(SEQL - kt - 32), DH, 32, DH);
                tdm_load_tile(Vh + (kt + 32),
                              (unsigned)(uintptr_t)&tV[buf ^ 1][0],
                              (unsigned)(SEQL - kt - 32), DH, 32, DH, SEQL);
                __builtin_amdgcn_s_wait_tensorcnt(2);
            } else {
                __builtin_amdgcn_s_wait_tensorcnt(0);
            }
        }
        __syncthreads();
#else
        {
            int t = threadIdx.x;
#pragma unroll
            for (int c = 0; c < (32 * DH) / (256 * 8); ++c) {
                int e = (c * 256 + t) * 8;
                int rK = e / DH,  cK = e % DH;
                *(v8bf*)&tK[buf][e] = *(const v8bf*)(Kh + (size_t)(kt + rK) * DH + cK);
                int rV = e / 32,  cV = e % 32;
                *(v8bf*)&tV[buf][e] = *(const v8bf*)(Vh + (size_t)rV * SEQL + kt + cV);
            }
        }
        __syncthreads();
#endif
        if (kt < qs + 16) {      // wave-uniform: this key block intersects range
            const __bf16* Kt = &tK[buf][0];
            const __bf16* Vtile = &tV[buf][0];
            v8f s0, s1;
#pragma unroll
            for (int e = 0; e < 8; ++e) { s0[e] = 0.0f; s1[e] = 0.0f; }
#pragma unroll
            for (int c = 0; c < 4; ++c) {
                v16bf b0 = load_frag(Kt, 0,  DH, c * 32, lane);
                s0 = wmma_bf16(aq[c], b0, s0);
                v16bf b1 = load_frag(Kt, 16, DH, c * 32, lane);
                s1 = wmma_bf16(aq[c], b1, s1);
            }

            float alpha[8];
#pragma unroll
            for (int i = 0; i < 8; ++i) {
                int q  = qs + i + 8 * hf;
                float v0 = s0[i] * scale;
                float v1 = s1[i] * scale;
                if (kt + nc > q)      v0 = -3.0e38f;
                if (kt + 16 + nc > q) v1 = -3.0e38f;
                float mx = fmaxf(v0, v1);
                mx = fmaxf(mx, __shfl_xor(mx, 1));
                mx = fmaxf(mx, __shfl_xor(mx, 2));
                mx = fmaxf(mx, __shfl_xor(mx, 4));
                mx = fmaxf(mx, __shfl_xor(mx, 8));
                float mnew = fmaxf(mi[i], mx);
                float a  = __expf(mi[i] - mnew);
                float p0 = __expf(v0 - mnew);
                float p1 = __expf(v1 - mnew);
                float rs = p0 + p1;
                rs += __shfl_xor(rs, 1);
                rs += __shfl_xor(rs, 2);
                rs += __shfl_xor(rs, 4);
                rs += __shfl_xor(rs, 8);
                li[i] = li[i] * a + rs;
                mi[i] = mnew;
                alpha[i] = a;
                lds_p[w][(i + 8 * hf) * 32 + nc]      = f2bf(p0);
                lds_p[w][(i + 8 * hf) * 32 + nc + 16] = f2bf(p1);
            }

#pragma unroll
            for (int j = 0; j < 8; ++j)
#pragma unroll
                for (int i = 0; i < 8; ++i) o[j][i] *= alpha[i];

            v16bf pa = load_frag(&lds_p[w][0], 0, 32, 0, lane);
#pragma unroll
            for (int j = 0; j < 8; ++j) {
                v16bf bv = load_frag(Vtile, j * 16, 32, 0, lane);
                o[j] = wmma_bf16(pa, bv, o[j]);
            }
        }
        __syncthreads();     // protect buf from next iteration's overwrite
        buf ^= 1;
    }

#pragma unroll
    for (int i = 0; i < 8; ++i) li[i] = 1.0f / li[i];
#pragma unroll
    for (int j = 0; j < 8; ++j)
#pragma unroll
        for (int e = 0; e < 8; ++e) {
            int q   = qs + e + 8 * hf;
            int col = h * DH + j * 16 + nc;
            Ob[(size_t)q * DM + col] = f2bf(o[j][e] * li[e]);
        }
}

extern "C" void kernel_launch(void* const* d_in, const int* in_sizes, int n_in,
                              void* d_out, int out_size, void* d_ws, size_t ws_size,
                              hipStream_t stream) {
    (void)in_sizes; (void)n_in; (void)out_size; (void)ws_size;
    const float* x  = (const float*)d_in[0];
    const float* Wq = (const float*)d_in[1];
    const float* bq = (const float*)d_in[2];
    const float* Wk = (const float*)d_in[3];
    const float* bk = (const float*)d_in[4];
    const float* Wv = (const float*)d_in[5];
    const float* bv = (const float*)d_in[6];
    const float* Wo = (const float*)d_in[7];
    const float* bo = (const float*)d_in[8];
    float* out = (float*)d_out;

    // workspace layout (240 MiB total)
    char* ws = (char*)d_ws;
    size_t off = 0;
    __bf16* xb  = (__bf16*)(ws + off); off += (size_t)SEQL * DM * 2;
    __bf16* wqb = (__bf16*)(ws + off); off += (size_t)DM * DM * 2;
    __bf16* wkb = (__bf16*)(ws + off); off += (size_t)DM * DM * 2;
    __bf16* wvb = (__bf16*)(ws + off); off += (size_t)DM * DM * 2;
    __bf16* wob = (__bf16*)(ws + off); off += (size_t)DM * DM * 2;
    float*  Pf  = (float*)(ws + off);  off += (size_t)SEQL * DM * 4;
    __bf16* qb  = (__bf16*)(ws + off); off += (size_t)SEQL * DM * 2;
    __bf16* kb  = (__bf16*)(ws + off); off += (size_t)SEQL * DM * 2;
    __bf16* vt  = (__bf16*)(ws + off); off += (size_t)SEQL * DM * 2;
    __bf16* ob  = (__bf16*)(ws + off); off += (size_t)SEQL * DM * 2;

    const int nx = SEQL * DM;
    const int nw = DM * DM;
    cvt_bf16_kernel<<<nx / 256, 256, 0, stream>>>(x,  xb,  nx);
    cvt_bf16_kernel<<<nw / 256, 256, 0, stream>>>(Wq, wqb, nw);
    cvt_bf16_kernel<<<nw / 256, 256, 0, stream>>>(Wk, wkb, nw);
    cvt_bf16_kernel<<<nw / 256, 256, 0, stream>>>(Wv, wvb, nw);
    cvt_bf16_kernel<<<nw / 256, 256, 0, stream>>>(Wo, wob, nw);

    dim3 ggrid(DM / 128, SEQL / 128);   // (32, 16)

    gemm_bf16_kernel<<<ggrid, 256, 0, stream>>>(xb, wqb, bq, Pf, SEQL, DM, DM);
    rope_pack_kernel<<<(SEQL * DM / 2) / 256, 256, 0, stream>>>(Pf, qb);
    gemm_bf16_kernel<<<ggrid, 256, 0, stream>>>(xb, wkb, bk, Pf, SEQL, DM, DM);
    rope_pack_kernel<<<(SEQL * DM / 2) / 256, 256, 0, stream>>>(Pf, kb);
    gemm_bf16_kernel<<<ggrid, 256, 0, stream>>>(xb, wvb, bv, Pf, SEQL, DM, DM);
    vt_pack_kernel<<<(SEQL * DM) / 256, 256, 0, stream>>>(Pf, vt);

    flash_attn_kernel<<<dim3(SEQL / 128, NH), 256, 0, stream>>>(qb, kb, vt, ob);

    gemm_bf16_kernel<<<ggrid, 256, 0, stream>>>(ob, wob, bo, out, SEQL, DM, DM);
}
